// MultiHeadAttention_17995912970901
// MI455X (gfx1250) — matrix operand order repaired and checked by reading, not compiled, not run on hardware
//
#include <hip/hip_runtime.h>
#include <hip/hip_bf16.h>

// MHA for MI455X (gfx1250). All matmuls via v_wmma_f32_16x16x32_bf16.
// Pipeline: 3x projection GEMM (f32 in -> bf16 [B,H,S,D]) -> flash attention
// (async-LDS-staged K tiles, transposed V tiles, online softmax) -> output GEMM.
// Workspace: 4 bf16 buffers of B*H*S*D = 4,194,304 elems -> 32 MB total.

#define N_HEADS 16
#define EMBED   1024
#define DK      64
#define BATCH   2
#define SEQ     2048

typedef __attribute__((ext_vector_type(16))) __bf16 v16bf;
typedef __attribute__((ext_vector_type(8)))  __bf16 v8bf;
typedef __attribute__((ext_vector_type(8)))  float  v8f;

union ABFrag { v16bf v; v8bf h[2]; __bf16 e[16]; };

// Load 8 contiguous f32 and truncate-convert to bf16 fragment elements.
__device__ __forceinline__ void cvt8(const float* __restrict__ p, __bf16* e) {
  float4 f0 = *(const float4*)p;
  float4 f1 = *(const float4*)(p + 4);
  e[0] = (__bf16)f0.x; e[1] = (__bf16)f0.y; e[2] = (__bf16)f0.z; e[3] = (__bf16)f0.w;
  e[4] = (__bf16)f1.x; e[5] = (__bf16)f1.y; e[6] = (__bf16)f1.z; e[7] = (__bf16)f1.w;
}

// Out[b,h,s,d] (bf16) = (X @ W^T)[m, n],  m = b*SEQ+s, n = h*DK+d.
// One 16x64 output strip per wave (A fragment reused across 4 WMMAs/k-step).
__global__ void __launch_bounds__(256)
mha_proj_kernel(const float* __restrict__ X, const float* __restrict__ W,
                __bf16* __restrict__ Out) {
  const int lane = threadIdx.x & 31;
  const int wave = threadIdx.x >> 5;
  const int half = lane >> 4;
  const int l16  = lane & 15;
  const int tile = blockIdx.x * 8 + wave;      // 4096 strips total
  const int ng   = EMBED / 64;                 // 16 column groups
  const int m0   = (tile / ng) * 16;
  const int n0   = (tile % ng) * 64;

  const float* aRow = X + (size_t)(m0 + l16) * EMBED;   // A: row m0+l16
  const float* bR   = W + (size_t)(n0 + l16) * EMBED;   // B[k][n] = W[n][k]

  v8f c[4] = {{}, {}, {}, {}};
  for (int k0 = 0; k0 < EMBED; k0 += 32) {
    ABFrag a;
    cvt8(aRow + k0 + half * 16,     a.e);     // A chunk0: K = k0+half*16 ..+7
    cvt8(aRow + k0 + half * 16 + 8, a.e + 8); // A chunk1: K = k0+half*16+8 ..+7
    #pragma unroll
    for (int t = 0; t < 4; ++t) {
      ABFrag b;
      cvt8(bR + (size_t)t * 16 * EMBED + k0 + half * 16,     b.e);
      cvt8(bR + (size_t)t * 16 * EMBED + k0 + half * 16 + 8, b.e + 8);
      c[t] = __builtin_amdgcn_wmma_f32_16x16x32_bf16(false, a.v, false, b.v,
                                                     (short)0, c[t], false, false);
    }
  }
  const int bb = m0 / SEQ;                 // whole 16-row tile in one batch
  const int s0 = (m0 % SEQ) + half * 8;    // C/D: row = r + half*8
  #pragma unroll
  for (int t = 0; t < 4; ++t) {
    const int n = n0 + t * 16 + l16;
    const int h = n >> 6, d = n & 63;
    size_t base = (((size_t)bb * N_HEADS + h) * SEQ + s0) * DK + d;
    #pragma unroll
    for (int r = 0; r < 8; ++r)
      Out[base + (size_t)r * DK] = (__bf16)c[t][r];
  }
}

// Flash attention over one (b,h): block = 8 waves x 16 query rows = 128 rows.
// K tiles staged via GLOBAL_LOAD_ASYNC_TO_LDS_B128 (ASYNCcnt); V transposed
// into LDS with ds stores; online softmax in registers.
__global__ void __launch_bounds__(256)
mha_attn_kernel(const __bf16* __restrict__ Qh, const __bf16* __restrict__ Kh,
                const __bf16* __restrict__ Vh,
                const unsigned char* __restrict__ mask,   // jnp bool: 1 byte/elem
                __bf16* __restrict__ Ctx) {
  __shared__ __align__(32) __bf16 sK[32 * DK];      // [key][d]       4 KB
  __shared__ __align__(32) __bf16 sVT[DK * 32];     // [d][key]       4 KB
  __shared__ __align__(32) __bf16 sP[8][16 * 32];   // per-wave P     8 KB

  const int tid  = threadIdx.x;
  const int lane = tid & 31, wave = tid >> 5;
  const int half = lane >> 4, l16 = lane & 15;
  const int bh = blockIdx.y;
  const int b  = bh / N_HEADS, h = bh % N_HEADS;
  const size_t headBase = ((size_t)b * N_HEADS + h) * SEQ * DK;
  const int q0 = blockIdx.x * 128 + wave * 16;
  const float scale = 0.125f;                        // 1/sqrt(DK)

  // Preload Q A-fragments: contraction d in [0,32) and [32,64).
  ABFrag aq0, aq1;
  {
    const __bf16* qrow = Qh + headBase + (size_t)(q0 + l16) * DK;
    aq0.h[0] = *(const v8bf*)(qrow + half * 16);
    aq0.h[1] = *(const v8bf*)(qrow + half * 16 + 8);
    aq1.h[0] = *(const v8bf*)(qrow + 32 + half * 16);
    aq1.h[1] = *(const v8bf*)(qrow + 32 + half * 16 + 8);
  }

  v8f acc0 = {}, acc1 = {}, acc2 = {}, acc3 = {};    // O tile 16x64
  float mrow[8], lrow[8];
  #pragma unroll
  for (int r = 0; r < 8; ++r) { mrow[r] = -3.0e38f; lrow[r] = 0.0f; }

  const int cidx = tid * 8;        // cooperative staging: 8 bf16/thread
  const int ckey = cidx >> 6;      // 0..31
  const int cd   = cidx & 63;

  // Uniform SGPR base for GVS-mode async loads of K; per-lane LDS dest addr.
  const unsigned long long kBase = (unsigned long long)(size_t)(Kh + headBase);
  const unsigned ldsK = (unsigned)(size_t)(&sK[cidx]);   // LDS byte offset

  for (int j0 = 0; j0 < SEQ; j0 += 32) {
    __syncthreads();
    // Stage K tile row-major: async global->LDS (16B per lane, ASYNCcnt).
    {
      unsigned voff = (unsigned)((j0 * DK + cidx) * sizeof(__bf16));
      asm volatile("global_load_async_to_lds_b128 %0, %1, %2"
                   :: "v"(ldsK), "v"(voff), "s"(kBase) : "memory");
    }
    // Stage V tile transposed: sVT[d][key].
    {
      uint4 raw = *(const uint4*)(Vh + headBase + (size_t)(j0 + ckey) * DK + cd);
      const __bf16* e = (const __bf16*)&raw;
      #pragma unroll
      for (int i = 0; i < 8; ++i) sVT[(cd + i) * 32 + ckey] = e[i];
    }
    asm volatile("s_wait_asynccnt 0x0" ::: "memory");
    __syncthreads();

    // Scores: 16 queries x 32 keys = two 16x16 WMMA tiles, 2 WMMAs each (d=64).
    float p[2][8];
    #pragma unroll
    for (int t = 0; t < 2; ++t) {
      const __bf16* krow = &sK[(t * 16 + l16) * DK];  // B col n = key t*16+l16
      v16bf b0 = *(const v16bf*)(krow + half * 16);        // d in [0,32)
      v16bf b1 = *(const v16bf*)(krow + 32 + half * 16);   // d in [32,64)
      v8f cs = {};
      cs = __builtin_amdgcn_wmma_f32_16x16x32_bf16(false, aq0.v, false, b0,
                                                   (short)0, cs, false, false);
      cs = __builtin_amdgcn_wmma_f32_16x16x32_bf16(false, aq1.v, false, b1,
                                                   (short)0, cs, false, false);
      const int kc = j0 + t * 16 + l16;
      #pragma unroll
      for (int r = 0; r < 8; ++r) {
        int qr = q0 + half * 8 + r;
        float v = cs[r] * scale;
        if (mask[((size_t)b * SEQ + qr) * SEQ + kc]) v = -1.0e9f;
        p[t][r] = v;
      }
    }

    // Online softmax across the 32 keys (reduce over 16-lane groups).
    #pragma unroll
    for (int r = 0; r < 8; ++r) {
      float vmax = fmaxf(p[0][r], p[1][r]);
      #pragma unroll
      for (int off = 8; off >= 1; off >>= 1)
        vmax = fmaxf(vmax, __shfl_xor(vmax, off, 16));
      float mnew  = fmaxf(mrow[r], vmax);
      float alpha = __expf(mrow[r] - mnew);
      float p0 = __expf(p[0][r] - mnew);
      float p1 = __expf(p[1][r] - mnew);
      float rs = p0 + p1;
      #pragma unroll
      for (int off = 8; off >= 1; off >>= 1)
        rs += __shfl_xor(rs, off, 16);
      lrow[r] = lrow[r] * alpha + rs;
      mrow[r] = mnew;
      acc0[r] *= alpha; acc1[r] *= alpha; acc2[r] *= alpha; acc3[r] *= alpha;
      // C-layout -> LDS (row-major 16x32) for A-layout re-read.
      sP[wave][(half * 8 + r) * 32 + l16]      = (__bf16)p0;
      sP[wave][(half * 8 + r) * 32 + 16 + l16] = (__bf16)p1;
    }

    // P(16x32) x V(32x64): A from per-wave LDS scratch, B from transposed V.
    ABFrag pa;
    pa.h[0] = *(const v8bf*)(&sP[wave][l16 * 32 + half * 16]);
    pa.h[1] = *(const v8bf*)(&sP[wave][l16 * 32 + half * 16 + 8]);
    {
      v16bf bv0 = *(const v16bf*)(&sVT[(0  + l16) * 32 + half * 16]);
      v16bf bv1 = *(const v16bf*)(&sVT[(16 + l16) * 32 + half * 16]);
      v16bf bv2 = *(const v16bf*)(&sVT[(32 + l16) * 32 + half * 16]);
      v16bf bv3 = *(const v16bf*)(&sVT[(48 + l16) * 32 + half * 16]);
      acc0 = __builtin_amdgcn_wmma_f32_16x16x32_bf16(false, pa.v, false, bv0,
                                                     (short)0, acc0, false, false);
      acc1 = __builtin_amdgcn_wmma_f32_16x16x32_bf16(false, pa.v, false, bv1,
                                                     (short)0, acc1, false, false);
      acc2 = __builtin_amdgcn_wmma_f32_16x16x32_bf16(false, pa.v, false, bv2,
                                                     (short)0, acc2, false, false);
      acc3 = __builtin_amdgcn_wmma_f32_16x16x32_bf16(false, pa.v, false, bv3,
                                                     (short)0, acc3, false, false);
    }
  }

  // Normalize and store context (bf16, [B,H,S,D]).
  #pragma unroll
  for (int r = 0; r < 8; ++r) {
    float rinv = 1.0f / lrow[r];
    size_t rowoff = headBase + (size_t)(q0 + half * 8 + r) * DK;
    Ctx[rowoff + 0  + l16] = (__bf16)(acc0[r] * rinv);
    Ctx[rowoff + 16 + l16] = (__bf16)(acc1[r] * rinv);
    Ctx[rowoff + 32 + l16] = (__bf16)(acc2[r] * rinv);
    Ctx[rowoff + 48 + l16] = (__bf16)(acc3[r] * rinv);
  }
}

// Out[m][n] (f32) = sum_e ctxflat[m][e] * W_O[n][e]; ctxflat gathered from
// head layout: e = h*DK + d  ->  Ctx[b][h][s][d]. 16x64 strip per wave.
__global__ void __launch_bounds__(256)
mha_out_kernel(const __bf16* __restrict__ Ctx, const float* __restrict__ W,
               float* __restrict__ Out) {
  const int lane = threadIdx.x & 31;
  const int wave = threadIdx.x >> 5;
  const int half = lane >> 4;
  const int l16  = lane & 15;
  const int tile = blockIdx.x * 8 + wave;
  const int ng   = EMBED / 64;
  const int m0   = (tile / ng) * 16;
  const int n0   = (tile % ng) * 64;

  const int arow = m0 + l16;
  const int ab = arow / SEQ, as = arow % SEQ;
  const size_t abase = ((size_t)ab * N_HEADS) * SEQ * DK + (size_t)as * DK;
  const float* bR = W + (size_t)(n0 + l16) * EMBED;

  v8f c[4] = {{}, {}, {}, {}};
  for (int k0 = 0; k0 < EMBED; k0 += 32) {
    ABFrag a;
    {
      int kk = k0 + half * 16;         // 8-elem chunk never crosses a head
      a.h[0] = *(const v8bf*)(Ctx + abase + (size_t)(kk >> 6) * (SEQ * DK) + (kk & 63));
      kk = k0 + half * 16 + 8;
      a.h[1] = *(const v8bf*)(Ctx + abase + (size_t)(kk >> 6) * (SEQ * DK) + (kk & 63));
    }
    #pragma unroll
    for (int t = 0; t < 4; ++t) {
      ABFrag bfr;
      cvt8(bR + (size_t)t * 16 * EMBED + k0 + half * 16,     bfr.e);
      cvt8(bR + (size_t)t * 16 * EMBED + k0 + half * 16 + 8, bfr.e + 8);
      c[t] = __builtin_amdgcn_wmma_f32_16x16x32_bf16(false, a.v, false, bfr.v,
                                                     (short)0, c[t], false, false);
    }
  }
  const size_t rowBase = (size_t)(m0 + half * 8) * EMBED + n0 + l16;
  #pragma unroll
  for (int t = 0; t < 4; ++t)
    #pragma unroll
    for (int r = 0; r < 8; ++r)
      Out[rowBase + (size_t)r * EMBED + t * 16] = c[t][r];
}

extern "C" void kernel_launch(void* const* d_in, const int* in_sizes, int n_in,
                              void* d_out, int out_size, void* d_ws, size_t ws_size,
                              hipStream_t stream) {
  (void)in_sizes; (void)n_in; (void)out_size; (void)ws_size;
  const float* Q    = (const float*)d_in[0];
  const float* K    = (const float*)d_in[1];
  const float* V    = (const float*)d_in[2];
  const unsigned char* mask = (const unsigned char*)d_in[3];  // bool array
  const float* W_Q  = (const float*)d_in[4];
  const float* W_K  = (const float*)d_in[5];
  const float* W_V  = (const float*)d_in[6];
  const float* W_O  = (const float*)d_in[7];
  float* out = (float*)d_out;

  const size_t HE = (size_t)BATCH * N_HEADS * SEQ * DK;  // 4,194,304 elems
  __bf16* qb = (__bf16*)d_ws;          // needs 4*HE*2 = 32 MB of workspace
  __bf16* kb = qb + HE;
  __bf16* vb = kb + HE;
  __bf16* cb = vb + HE;

  dim3 blk(256);
  const int gemmBlocks = ((BATCH * SEQ) / 16) * (EMBED / 64) / 8;  // 512
  mha_proj_kernel<<<gemmBlocks, blk, 0, stream>>>(Q, W_Q, qb);
  mha_proj_kernel<<<gemmBlocks, blk, 0, stream>>>(K, W_K, kb);
  mha_proj_kernel<<<gemmBlocks, blk, 0, stream>>>(V, W_V, vb);
  mha_attn_kernel<<<dim3(SEQ / 128, BATCH * N_HEADS), blk, 0, stream>>>(
      qb, kb, vb, mask, cb);
  mha_out_kernel<<<gemmBlocks, blk, 0, stream>>>(cb, W_O, out);
}
